// RNNnet_69337952026981
// MI455X (gfx1250) — matrix operand order repaired and checked by reading, not compiled, not run on hardware
//
#include <hip/hip_runtime.h>

#define SEQ    8192
#define BATCH  512
#define HID    18
#define CH     64          // x-streaming chunk (timesteps)
#define LOGCH  6
#define NBUF   4           // quad buffer: WAR-safe by construction

typedef __attribute__((ext_vector_type(16))) _Float16 v16h;
typedef __attribute__((ext_vector_type(8)))  _Float16 v8h;
typedef __attribute__((ext_vector_type(8)))  float    v8f;
typedef __attribute__((ext_vector_type(4)))  float    v4f;
typedef int v4i __attribute__((vector_size(16)));

// tanh(x) = 1 - 2/(exp2(2*log2(e)*x) + 1); exact saturation at +-1.
__device__ __forceinline__ float fast_tanh(float x) {
    float e = __builtin_amdgcn_exp2f(x * 2.885390081777927f);
    return 1.0f - 2.0f * __builtin_amdgcn_rcpf(e + 1.0f);
}

#if __has_builtin(__builtin_amdgcn_tanhf)
#define TANHF(v) __builtin_amdgcn_tanhf(v)     // v_tanh_f32: 1 trans op
#else
#define TANHF(v) fast_tanh(v)
#endif

#if __has_builtin(__builtin_amdgcn_global_load_async_to_lds_b128)
#define HAVE_ASYNC_LDS 1
typedef __attribute__((address_space(1))) v4i GV4;   // global 16B vector
typedef __attribute__((address_space(3))) v4i LV4;   // LDS 16B vector
#else
#define HAVE_ASYNC_LDS 0
#endif

__device__ __forceinline__ void wait_async8() {
#if HAVE_ASYNC_LDS
#if __has_builtin(__builtin_amdgcn_s_wait_asynccnt)
    __builtin_amdgcn_s_wait_asynccnt(8);
#else
    asm volatile("s_wait_asynccnt 8" ::: "memory");
#endif
#endif
    asm volatile("" ::: "memory");   // compiler fence: LDS now holds new chunk
}

// Copy chunk tc of x (CH rows x 64B) into xb[tc % NBUF]. 8 x 16B per lane.
__device__ __forceinline__ void fill_chunk(const float* __restrict__ x,
                                           float* __restrict__ xb,
                                           int b0, int lane, int tc) {
    float* dst = xb + (tc & (NBUF - 1)) * (CH * 16);
#pragma unroll
    for (int i = 0; i < 8; ++i) {
        const int idx16 = i * 32 + lane;        // 16B unit within chunk
        const int row   = idx16 >> 2;           // timestep within chunk
        const int col   = (idx16 & 3) * 4;      // float offset within 64B row
        int trow = tc * CH + row;
        if (trow > SEQ - 1) trow = SEQ - 1;     // clamp trailing prefetch
        const float* g = x + (size_t)trow * BATCH + b0 + col;
        float*       l = dst + row * 16 + col;
#if HAVE_ASYNC_LDS
        __builtin_amdgcn_global_load_async_to_lds_b128((GV4*)g, (LV4*)l, 0, 0);
#else
        *(v4f*)l = *(const v4f*)g;              // synchronous fallback
#endif
    }
}

// One wave owns one 16-row batch tile; full 8192-step scan via
// V_WMMA_F32_16X16X32_F16.  A = [16 batch x K32]: K0..17 = h_{t-1},
// K18 = x_t, K19 = 1.0 (bias slot), K20..31 = 0.  B rows: W_hh^T; W_ih; b_ih+b_hh.
__global__ __launch_bounds__(32) void rnn_scan_kernel(
    const float* __restrict__ x,     const float* __restrict__ W_ih,
    const float* __restrict__ b_ih,  const float* __restrict__ W_hh,
    const float* __restrict__ b_hh,  const float* __restrict__ W_out,
    const float* __restrict__ b_out, float* __restrict__ out)
{
    __shared__ __align__(16) _Float16 hbuf[2][16 * 32];   // double-buffered h tile
    __shared__ __align__(16) float    xb[NBUF * CH * 16]; // x stream buffers (16KB)

    const int lane = threadIdx.x;
    const int half = lane >> 4;
    const int n    = lane & 15;
    const int b0   = blockIdx.x * 16;

    // zero h tiles: padding columns must read back 0 forever
    for (int i = lane; i < 2 * 16 * 32; i += 32) hbuf[0][i] = (_Float16)0.0f;
    __syncthreads();

    // ---- B operands. B layout (32x16 KxN f16): e -> K=(e&7)+16*(e>>3)+8*(lane/16)
    v16h B0, B1;
#pragma unroll
    for (int e = 0; e < 16; ++e) {
        const int k  = (e & 7) + 16 * (e >> 3) + 8 * half;
        const int j1 = 16 + n;
        float v0 = 0.0f, v1 = 0.0f;
        if (k < HID) {
            v0 = W_hh[n * HID + k];
            if (j1 < HID) v1 = W_hh[j1 * HID + k];
        } else if (k == HID) {            // x slot
            v0 = W_ih[n];
            if (j1 < HID) v1 = W_ih[j1];
        } else if (k == HID + 1) {        // bias slot
            v0 = b_ih[n] + b_hh[n];
            if (j1 < HID) v1 = b_ih[j1] + b_hh[j1];
        }
        B0[e] = (_Float16)v0;
        B1[e] = (_Float16)v1;
    }

    // W_out gathered into A layout: e -> K = (e&7) + 16*(e>>3) + 8*half
    float wv[16];
#pragma unroll
    for (int e = 0; e < 16; ++e) {
        const int K = (e & 7) + 16 * (e >> 3) + 8 * half;
        wv[e] = (K < HID) ? W_out[K] : 0.0f;
    }
    const float bo = b_out[0];

    // ---- prime x stream: chunks 0 and 1 in flight, wait for chunk 0
    fill_chunk(x, xb, b0, lane, 0);
    fill_chunk(x, xb, b0, lane, 1);
    wait_async8();

    // ---- A state: h_{-1}=0, seed x_0 (K=18 -> e=10, lanes<16) and 1.0 (K=19 -> e=11)
    v16h A;
#pragma unroll
    for (int e = 0; e < 16; ++e) A[e] = (_Float16)0.0f;
    {
        const float x0 = xb[n];               // chunk 0, row 0
        if (!half) { A[10] = (_Float16)x0; A[11] = (_Float16)1.0f; }
    }

    v8f czero;
#pragma unroll
    for (int r = 0; r < 8; ++r) czero[r] = 0.0f;

    const int rb = 8 * half;

#pragma unroll 2
    for (int t = 0; t < SEQ; ++t) {
        // chunk boundary: next step enters a new chunk -> prefetch +1, wait current
        if (((t + 1) & (CH - 1)) == 0) {
            fill_chunk(x, xb, b0, lane, ((t + 1) >> LOGCH) + 1);
            wait_async8();
        }

        // pre-activation: D = A x B + 0 (f32 accum)
        v8f d0 = __builtin_amdgcn_wmma_f32_16x16x32_f16(
                     false, A, false, B0, (short)0, czero, false, false);
        v8f d1 = __builtin_amdgcn_wmma_f32_16x16x32_f16(
                     false, A, false, B1, (short)0, czero, false, false);

        // tanh + scatter h_t to LDS tile (b-major, 32-half pitch)
        _Float16* hb = hbuf[t & 1];
#pragma unroll
        for (int r = 0; r < 8; ++r)
            hb[(rb + r) * 32 + n] = (_Float16)TANHF(d0[r]);
        if (n < 2) {                           // columns j = 16,17
#pragma unroll
            for (int r = 0; r < 8; ++r)
                hb[(rb + r) * 32 + 16 + n] = (_Float16)TANHF(d1[r]);
        }
        __syncthreads();                       // single-wave WG: ~free LDS fence

        // rebuild A layout: two contiguous 16B LDS loads
        const _Float16* base = &hb[n * 32 + 8 * half];
        v8h lo = *(const v8h*)base;
        v8h hi = *(const v8h*)(base + 16);
#pragma unroll
        for (int e = 0; e < 8; ++e) { A[e] = lo[e]; A[e + 8] = hi[e]; }

        // x_{t+1} from the LDS stream
        const int tn = t + 1;
        const float xnext =
            xb[((tn >> LOGCH) & (NBUF - 1)) * (CH * 16) + (tn & (CH - 1)) * 16 + n];
        if (!half) { A[10] = (_Float16)xnext; A[11] = (_Float16)1.0f; }

        // o_t = h_t . W_out + b_out   (wv zero at x/bias slots; off critical path)
        float p = 0.0f;
#pragma unroll
        for (int e = 0; e < 16; ++e) p += (float)A[e] * wv[e];
        p += __shfl_xor(p, 16, 32);
        if (!half) out[(size_t)t * BATCH + b0 + n] = p + bo;
    }

    // final hidden state -> out[SEQ*BATCH + b*HID + j]
    float* hid = out + (size_t)SEQ * BATCH;
#pragma unroll
    for (int e = 0; e < 16; ++e) {
        const int K = (e & 7) + 16 * (e >> 3) + 8 * half;
        if (K < HID) hid[(size_t)(b0 + n) * HID + K] = (float)A[e];
    }
}

extern "C" void kernel_launch(void* const* d_in, const int* in_sizes, int n_in,
                              void* d_out, int out_size, void* d_ws, size_t ws_size,
                              hipStream_t stream) {
    (void)in_sizes; (void)n_in; (void)out_size; (void)d_ws; (void)ws_size;
    const float* x     = (const float*)d_in[0];
    const float* W_ih  = (const float*)d_in[1];
    const float* b_ih  = (const float*)d_in[2];
    const float* W_hh  = (const float*)d_in[3];
    const float* b_hh  = (const float*)d_in[4];
    const float* W_out = (const float*)d_in[5];
    const float* b_out = (const float*)d_in[6];
    float* out = (float*)d_out;

    rnn_scan_kernel<<<dim3(BATCH / 16), dim3(32), 0, stream>>>(
        x, W_ih, b_ih, W_hh, b_hh, W_out, b_out, out);
}
